// W8SDLinear_34883724378157
// MI455X (gfx1250) — 8-bit matrix operand order repaired and checked by reading, compiled, not run on hardware
//
#include <hip/hip_runtime.h>

#define DIN   4096
#define DOUT  4096
#define MTOK  8192
#define KOUT  16
#define KB64  (DIN / 64)
#define NB16  (DOUT / 16)

typedef int v8i __attribute__((ext_vector_type(8)));

#if defined(__has_builtin)
#if __has_builtin(__builtin_amdgcn_s_wait_asynccnt)
#define WAIT_ASYNC(n) __builtin_amdgcn_s_wait_asynccnt(n)
#endif
#endif
#ifndef WAIT_ASYNC
#define WAIT_ASYNC(n) asm volatile("s_wait_asynccnt %0" ::"i"(n) : "memory")
#endif

__device__ __forceinline__ unsigned lds_off(const void* p) {
    return (unsigned)(unsigned long long)(__attribute__((address_space(3))) const void*)p;
}

__device__ __forceinline__ void async_ld16(unsigned lds_addr, const void* gsrc) {
    asm volatile("global_load_async_to_lds_b128 %0, %1, off"
                 :: "v"(lds_addr), "v"(gsrc)
                 : "memory");
}

__global__ void topk_kernel(const float* __restrict__ act_max, int* __restrict__ idx_out) {
    __shared__ float sv[DIN];
    __shared__ float rv[256];
    __shared__ int   ri[256];
    int t = threadIdx.x;
    for (int k = t; k < DIN; k += 256) sv[k] = fabsf(act_max[k]);
    __syncthreads();
    for (int sel = 0; sel < KOUT; ++sel) {
        float bv = -1.0f; int bi = 0;
        for (int k = t; k < DIN; k += 256) {
            float v = sv[k];
            if (v > bv) { bv = v; bi = k; }
        }
        rv[t] = bv; ri[t] = bi;
        __syncthreads();
        if (t == 0) {
            float Bv = -1.0f; int Bi = 0;
            for (int i = 0; i < 256; ++i) {
                if (rv[i] > Bv || (rv[i] == Bv && ri[i] < Bi)) { Bv = rv[i]; Bi = ri[i]; }
            }
            idx_out[sel] = Bi;
            sv[Bi] = -2.0f;
        }
        __syncthreads();
    }
}

__global__ void wquant_kernel(const float* __restrict__ weight, const int* __restrict__ idx,
                              signed char* __restrict__ wq, _Float16* __restrict__ wmax,
                              float* __restrict__ wout) {
    int n = blockIdx.x;
    int t = threadIdx.x;
    __shared__ int   sidx[KOUT];
    __shared__ float red[256];
    if (t < KOUT) sidx[t] = idx[t];
    __syncthreads();

    const float* row = weight + (size_t)n * DIN;
    float vals[16];
    int k0 = t * 16;
    #pragma unroll
    for (int i = 0; i < 4; ++i) {
        float4 v = *(const float4*)(row + k0 + i * 4);
        vals[i*4+0] = v.x; vals[i*4+1] = v.y; vals[i*4+2] = v.z; vals[i*4+3] = v.w;
    }
    float amax = 0.0f;
    #pragma unroll
    for (int i = 0; i < 16; ++i) {
        int k = k0 + i;
        bool z = false;
        #pragma unroll
        for (int j = 0; j < KOUT; ++j) z |= (k == sidx[j]);
        if (z) vals[i] = 0.0f;
        amax = fmaxf(amax, fabsf(vals[i]));
    }
    red[t] = amax; __syncthreads();
    for (int s = 128; s > 0; s >>= 1) {
        if (t < s) red[t] = fmaxf(red[t], red[t + s]);
        __syncthreads();
    }
    _Float16 sh = (_Float16)(red[0] / 127.0f);
    if (t == 0) wmax[n] = sh;
    float s = (float)sh;
    int nb = n >> 4, nin = n & 15;
    #pragma unroll
    for (int i = 0; i < 16; ++i) {
        int k = k0 + i;
        int q = (int)truncf(vals[i] / s);
        q = q > 127 ? 127 : (q < -128 ? -128 : q);
        int kb = k >> 6, kin = k & 63;
        int lane = nin + 16 * ((kin >> 3) & 1);
        int half = (kin >> 4) * 8 + (kin & 7);
        size_t off = ((size_t)(kb * NB16 + nb) * 32 + lane) * 32 + half;
        wq[off] = (signed char)q;
    }
    if (t < KOUT) wout[(size_t)t * DOUT + n] = row[sidx[t]];
}

__global__ void xquant_kernel(const float* __restrict__ x, const int* __restrict__ idx,
                              signed char* __restrict__ xq, _Float16* __restrict__ xmax,
                              float* __restrict__ xout) {
    int m = blockIdx.x;
    int t = threadIdx.x;
    __shared__ int   sidx[KOUT];
    __shared__ float red[256];
    if (t < KOUT) sidx[t] = idx[t];
    __syncthreads();

    const float* row = x + (size_t)m * DIN;
    float vals[16];
    int k0 = t * 16;
    #pragma unroll
    for (int i = 0; i < 4; ++i) {
        float4 v = *(const float4*)(row + k0 + i * 4);
        vals[i*4+0] = v.x; vals[i*4+1] = v.y; vals[i*4+2] = v.z; vals[i*4+3] = v.w;
    }
    float amax = 0.0f;
    #pragma unroll
    for (int i = 0; i < 16; ++i) {
        int k = k0 + i;
        bool z = false;
        #pragma unroll
        for (int j = 0; j < KOUT; ++j) z |= (k == sidx[j]);
        if (z) vals[i] = 0.0f;
        amax = fmaxf(amax, fabsf(vals[i]));
    }
    red[t] = amax; __syncthreads();
    for (int s = 128; s > 0; s >>= 1) {
        if (t < s) red[t] = fmaxf(red[t], red[t + s]);
        __syncthreads();
    }
    _Float16 sh = (_Float16)(red[0] / 127.0f);
    if (t == 0) xmax[m] = sh;
    float s = (float)sh;
    int mb = m >> 4, min_ = m & 15;
    #pragma unroll
    for (int i = 0; i < 16; ++i) {
        int k = k0 + i;
        int q = (int)truncf(vals[i] / s);
        q = q > 127 ? 127 : (q < -128 ? -128 : q);
        int kb = k >> 6, kin = k & 63;
        int lane = min_ + 16 * ((kin >> 3) & 1);
        int byte = (kin >> 4) * 8 + (kin & 7);
        size_t off = ((size_t)(mb * KB64 + kb) * 32 + lane) * 32 + byte;
        xq[off] = (signed char)q;
    }
    if (t < KOUT) xout[(size_t)m * KOUT + t] = row[sidx[t]];
}

__global__ void __launch_bounds__(256)
gemm_kernel(const signed char* __restrict__ aP, const signed char* __restrict__ bP,
            const _Float16* __restrict__ xmax, const _Float16* __restrict__ wmax,
            const float* __restrict__ xout, const float* __restrict__ wout,
            const float* __restrict__ bias, float* __restrict__ out) {
    __shared__ __align__(16) char sA[2][8192];
    __shared__ __align__(16) char sB[2][8192];

    int t = threadIdx.x;
    int lane = t & 31;
    int wave = t >> 5;
    int waveRow = wave >> 1;
    int waveCol = wave & 1;
    int mb0  = blockIdx.y * 8 + waveRow * 2;
    int nb0g = blockIdx.x * 8 + waveCol * 4;

    int rA = t >> 5;
    int wA = (t & 31) * 32;
    const signed char* aNext = aP + ((size_t)(blockIdx.y * 8 + rA) * KB64) * 1024 + wA;
    const signed char* bNext = bP + ((size_t)blockIdx.x * 8) * 1024 + (size_t)t * 32;

    const char* sAp = &sA[0][0];
    const char* sBp = &sB[0][0];
    unsigned la = lds_off(sAp) + (unsigned)(t * 32);
    unsigned lb = lds_off(sBp) + (unsigned)(t * 32);

    int rowBase = waveRow * 2048 + lane * 32;
    int colBase = waveCol * 4096 + lane * 32;

    v8i zero = {0, 0, 0, 0, 0, 0, 0, 0};
    v8i acc[2][4];
    #pragma unroll
    for (int i = 0; i < 2; ++i)
        #pragma unroll
        for (int j = 0; j < 4; ++j) acc[i][j] = zero;

    async_ld16(la, aNext);          async_ld16(la + 16, aNext + 16);
    async_ld16(lb, bNext);          async_ld16(lb + 16, bNext + 16);
    aNext += 1024;
    bNext += (size_t)NB16 * 1024;

    unsigned curOff = 0;
    #pragma unroll 1
    for (int kb = 0; kb < KB64 - 1; ++kb) {
        unsigned nxtOff = curOff ^ 8192u;
        async_ld16(la + nxtOff, aNext);  async_ld16(la + nxtOff + 16, aNext + 16);
        async_ld16(lb + nxtOff, bNext);  async_ld16(lb + nxtOff + 16, bNext + 16);
        aNext += 1024;
        bNext += (size_t)NB16 * 1024;

        WAIT_ASYNC(4);
        __syncthreads();

        const char* Ab = sAp + curOff + rowBase;
        const char* Bb = sBp + curOff + colBase;
        v8i a0 = *(const v8i*)(Ab);
        v8i a1 = *(const v8i*)(Ab + 1024);
        v8i b0 = *(const v8i*)(Bb);
        v8i b1 = *(const v8i*)(Bb + 1024);
        v8i b2 = *(const v8i*)(Bb + 2048);
        v8i b3 = *(const v8i*)(Bb + 3072);
        acc[0][0] = __builtin_amdgcn_wmma_i32_16x16x64_iu8(true, a0, true, b0, acc[0][0], false, false);
        acc[1][0] = __builtin_amdgcn_wmma_i32_16x16x64_iu8(true, a1, true, b0, acc[1][0], false, false);
        acc[0][1] = __builtin_amdgcn_wmma_i32_16x16x64_iu8(true, a0, true, b1, acc[0][1], false, false);
        acc[1][1] = __builtin_amdgcn_wmma_i32_16x16x64_iu8(true, a1, true, b1, acc[1][1], false, false);
        acc[0][2] = __builtin_amdgcn_wmma_i32_16x16x64_iu8(true, a0, true, b2, acc[0][2], false, false);
        acc[1][2] = __builtin_amdgcn_wmma_i32_16x16x64_iu8(true, a1, true, b2, acc[1][2], false, false);
        acc[0][3] = __builtin_amdgcn_wmma_i32_16x16x64_iu8(true, a0, true, b3, acc[0][3], false, false);
        acc[1][3] = __builtin_amdgcn_wmma_i32_16x16x64_iu8(true, a1, true, b3, acc[1][3], false, false);

        __syncthreads();
        curOff = nxtOff;
    }
    WAIT_ASYNC(0);
    __syncthreads();
    {
        const char* Ab = sAp + curOff + rowBase;
        const char* Bb = sBp + curOff + colBase;
        v8i a0 = *(const v8i*)(Ab);
        v8i a1 = *(const v8i*)(Ab + 1024);
        v8i b0 = *(const v8i*)(Bb);
        v8i b1 = *(const v8i*)(Bb + 1024);
        v8i b2 = *(const v8i*)(Bb + 2048);
        v8i b3 = *(const v8i*)(Bb + 3072);
        acc[0][0] = __builtin_amdgcn_wmma_i32_16x16x64_iu8(true, a0, true, b0, acc[0][0], false, false);
        acc[1][0] = __builtin_amdgcn_wmma_i32_16x16x64_iu8(true, a1, true, b0, acc[1][0], false, false);
        acc[0][1] = __builtin_amdgcn_wmma_i32_16x16x64_iu8(true, a0, true, b1, acc[0][1], false, false);
        acc[1][1] = __builtin_amdgcn_wmma_i32_16x16x64_iu8(true, a1, true, b1, acc[1][1], false, false);
        acc[0][2] = __builtin_amdgcn_wmma_i32_16x16x64_iu8(true, a0, true, b2, acc[0][2], false, false);
        acc[1][2] = __builtin_amdgcn_wmma_i32_16x16x64_iu8(true, a1, true, b2, acc[1][2], false, false);
        acc[0][3] = __builtin_amdgcn_wmma_i32_16x16x64_iu8(true, a0, true, b3, acc[0][3], false, false);
        acc[1][3] = __builtin_amdgcn_wmma_i32_16x16x64_iu8(true, a1, true, b3, acc[1][3], false, false);
    }

    int nlane = lane & 15;
    int rhalf = lane >> 4;
    #pragma unroll
    for (int j = 0; j < 4; ++j) {
        int col = (nb0g + j) * 16 + nlane;
        _Float16 wmh = wmax[col];
        float bi = bias[col];
        float wo[KOUT];
        #pragma unroll
        for (int jj = 0; jj < KOUT; ++jj) wo[jj] = wout[(size_t)jj * DOUT + col];
        #pragma unroll
        for (int i = 0; i < 2; ++i) {
            int mBase = (mb0 + i) * 16 + rhalf * 8;
            #pragma unroll
            for (int r = 0; r < 8; ++r) {
                int row = mBase + r;
                _Float16 prod = xmax[row] * wmh;
                float v = (float)acc[i][j][r] * (float)prod;
                const float* xo = xout + (size_t)row * KOUT;
                float s = bi;
                #pragma unroll
                for (int jj = 0; jj < KOUT; ++jj) s += xo[jj] * wo[jj];
                out[(size_t)row * DOUT + col] = v + s;
            }
        }
    }
}

extern "C" void kernel_launch(void* const* d_in, const int* in_sizes, int n_in,
                              void* d_out, int out_size, void* d_ws, size_t ws_size,
                              hipStream_t stream) {
    const float* x       = (const float*)d_in[0];
    const float* weight  = (const float*)d_in[1];
    const float* bias    = (const float*)d_in[2];
    const float* act_max = (const float*)d_in[3];
    float* out = (float*)d_out;

    char* ws = (char*)d_ws;
    size_t off = 0;
    auto carve = [&](size_t bytes) -> void* {
        void* p = ws + off;
        off = (off + bytes + 255) & ~(size_t)255;
        return p;
    };
    int*         idx   = (int*)carve(KOUT * sizeof(int));
    _Float16*    wmax  = (_Float16*)carve(DOUT * sizeof(_Float16));
    _Float16*    xmax  = (_Float16*)carve(MTOK * sizeof(_Float16));
    float*       wout  = (float*)carve((size_t)KOUT * DOUT * sizeof(float));
    float*       xout  = (float*)carve((size_t)MTOK * KOUT * sizeof(float));
    signed char* wq    = (signed char*)carve((size_t)DOUT * DIN);
    signed char* xq    = (signed char*)carve((size_t)MTOK * DIN);
    (void)ws_size; (void)in_sizes; (void)n_in; (void)out_size;

    topk_kernel<<<1, 256, 0, stream>>>(act_max, idx);
    wquant_kernel<<<DOUT, 256, 0, stream>>>(weight, idx, wq, wmax, wout);
    xquant_kernel<<<MTOK, 256, 0, stream>>>(x, idx, xq, xmax, xout);

    dim3 grid(DOUT / 128, MTOK / 128, 1);
    gemm_kernel<<<grid, 256, 0, stream>>>(xq, wq, xmax, wmax, xout, wout, bias, out);
}
